// OnlineLabeler_24060406792368
// MI455X (gfx1250) — matrix operand order repaired and checked by reading, not compiled, not run on hardware
//
#include <hip/hip_runtime.h>
#include <hip/hip_bf16.h>

// ---------------------------------------------------------------------------
// Problem constants (from reference): features [N,D] f32, ck [CT,D] f32,
// current_classes [K] int. Output: preds [N] (written as float per harness).
// ---------------------------------------------------------------------------
#define NROWS   16384
#define DDIM    1024
#define KCLS    512

typedef _Float16 v16h __attribute__((ext_vector_type(16)));
typedef _Float16 h8   __attribute__((ext_vector_type(8)));
typedef _Float16 h4   __attribute__((ext_vector_type(4)));
typedef float    v8f  __attribute__((ext_vector_type(8)));
typedef int      v4i  __attribute__((ext_vector_type(4)));

#define GLOBAL_AS __attribute__((address_space(1)))
#define LDS_AS    __attribute__((address_space(3)))

#if defined(__has_builtin)
#  if __has_builtin(__builtin_amdgcn_global_load_async_to_lds_b128)
#    define HAVE_ASYNC_LDS 1
#  else
#    define HAVE_ASYNC_LDS 0
#  endif
#  if __has_builtin(__builtin_amdgcn_s_wait_asynccnt)
#    define HAVE_WAIT_ASYNC 1
#  else
#    define HAVE_WAIT_ASYNC 0
#  endif
#else
#  define HAVE_ASYNC_LDS 0
#  define HAVE_WAIT_ASYNC 0
#endif

__device__ __forceinline__ void cp16_g2l(void* lds_dst, const void* gsrc) {
#if HAVE_ASYNC_LDS
  __builtin_amdgcn_global_load_async_to_lds_b128(
      (GLOBAL_AS v4i*)const_cast<void*>(gsrc),
      (LDS_AS v4i*)lds_dst, 0, 0);
#else
  *(h8*)lds_dst = *(const h8*)gsrc;
#endif
}

__device__ __forceinline__ void async_copies_wait() {
#if HAVE_ASYNC_LDS
#  if HAVE_WAIT_ASYNC
  __builtin_amdgcn_s_wait_asynccnt(0);
#  else
  asm volatile("s_wait_asynccnt 0" ::: "memory");
#  endif
#endif
}

// ---------------------------------------------------------------------------
// Kernel 1/2: L2-normalize a row of 1024 f32 and emit f16.
// useIdx=0: row = blockIdx.x (features). useIdx=1: row = idx[blockIdx.x] (ck
// gather). One block (256 threads) per output row.
// ---------------------------------------------------------------------------
__global__ __launch_bounds__(256)
void ol_norm_rows_f16(const float* __restrict__ src, _Float16* __restrict__ dst,
                      const int* __restrict__ idx, int useIdx) {
  const int tid  = threadIdx.x;
  const int row  = blockIdx.x;
  const int srow = useIdx ? idx[row] : row;

  const float4 v = ((const float4*)(src + (size_t)srow * DDIM))[tid];
  float s = v.x * v.x + v.y * v.y + v.z * v.z + v.w * v.w;
  #pragma unroll
  for (int off = 16; off > 0; off >>= 1) s += __shfl_down(s, off, 32);

  __shared__ float part[8];
  if ((tid & 31) == 0) part[tid >> 5] = s;
  __syncthreads();
  float tot = 0.f;
  #pragma unroll
  for (int i = 0; i < 8; ++i) tot += part[i];

  const float scale = 1.0f / fmaxf(sqrtf(tot), 1e-12f);
  h4 o;
  o[0] = (_Float16)(v.x * scale);
  o[1] = (_Float16)(v.y * scale);
  o[2] = (_Float16)(v.z * scale);
  o[3] = (_Float16)(v.w * scale);
  *(h4*)(dst + (size_t)row * DDIM + tid * 4) = o;
}

// ---------------------------------------------------------------------------
// Kernel 3: fused WMMA GEMM (sims = Af16 @ Bf16^T) + per-row argmax + class
// gather.
//   Block  = 256 threads (8 wave32); output band = 64 rows.
//   kt loop: 4 column tiles of BN=128 classes.
//   dk loop: 16 chunks of BD=64 along D, double-buffered LDS staged with
//            global_load_async_to_lds_b128 (next chunk overlaps WMMA).
//   Wave w: rows (w&1)*32 + {0,16}, cols (w>>1)*32 + {0,16}  -> 2x2 subtiles,
//           4 accumulators, 2 A frags + 2 B frags per K-step (2 ds/wmma).
//   Argmax merged via packed (orderable-f32 << 32 | col) u64 LDS atomicMax.
// ---------------------------------------------------------------------------
#define BM 64
#define BN 128
#define BD 64
#define LDA (BD + 8)        // +16B pad: row stride 144B spreads ds banks
#define NCHUNK (DDIM / BD)  // 16

__device__ __forceinline__ void stage_chunk(
    const _Float16* __restrict__ A, const _Float16* __restrict__ B,
    _Float16 (*As)[LDA], _Float16 (*Bs)[LDA],
    int tid, int m0, int c0, int dk) {
  // A tile: 64 rows x 128B = 512 x 16B units (2 per thread), fully unrolled.
  #pragma unroll
  for (int k = 0; k < 2; ++k) {
    const int u = tid + k * 256;
    const int r = u >> 3, cu = (u & 7) * 8;
    cp16_g2l(&As[r][cu], &A[(size_t)(m0 + r) * DDIM + dk + cu]);
  }
  // B tile: 128 rows x 128B = 1024 x 16B units (4 per thread).
  #pragma unroll
  for (int k = 0; k < 4; ++k) {
    const int u = tid + k * 256;
    const int r = u >> 3, cu = (u & 7) * 8;
    cp16_g2l(&Bs[r][cu], &B[(size_t)(c0 + r) * DDIM + dk + cu]);
  }
}

__device__ __forceinline__ void argmax_merge(unsigned long long* bestKey,
                                             int row, float v, int c) {
  unsigned ub = __float_as_uint(v);
  ub = (ub & 0x80000000u) ? ~ub : (ub | 0x80000000u);  // orderable f32 map
  atomicMax(&bestKey[row],
            ((unsigned long long)ub << 32) | (unsigned)c);  // -> ds_max_u64
}

__global__ __launch_bounds__(256)
void ol_gemm_argmax(const _Float16* __restrict__ A, const _Float16* __restrict__ B,
                    const int* __restrict__ cls, float* __restrict__ out) {
  __shared__ _Float16 As[2][BM][LDA];
  __shared__ _Float16 Bs[2][BN][LDA];
  __shared__ unsigned long long bestKey[BM];

  const int tid  = threadIdx.x;
  const int lane = tid & 31;
  const int wv   = tid >> 5;
  const int m0   = blockIdx.x * BM;

  const int wr = (wv & 1) * 32;   // wave row-block base (2x16 rows)
  const int wc = (wv >> 1) * 32;  // wave col-block base (2x16 cols)
  const int lr = lane & 15;
  const int hi = lane >> 4;       // lane-half select (ISA 7.12.2 layouts)

  if (tid < BM) bestKey[tid] = 0ull;
  __syncthreads();

  for (int kt = 0; kt < KCLS / BN; ++kt) {
    const int c0 = kt * BN;
    v8f acc00 = {}, acc01 = {}, acc10 = {}, acc11 = {};

    // Prologue: stage chunk 0 into buffer 0.
    stage_chunk(A, B, As[0], Bs[0], tid, m0, c0, 0);
    async_copies_wait();
    __syncthreads();

    for (int i = 0; i < NCHUNK; ++i) {
      const int cur = i & 1;
      // Overlap: issue async copies for chunk i+1 into the other buffer
      // while this chunk's WMMAs run.
      if (i + 1 < NCHUNK)
        stage_chunk(A, B, As[cur ^ 1], Bs[cur ^ 1], tid, m0, c0, (i + 1) * BD);

      #pragma unroll
      for (int kk = 0; kk < BD; kk += 32) {
        // A fragments (16x32 f16): lane<16 holds K 0..7 / 16..23, lane>=16
        // holds K 8..15 / 24..31 of its row -> two 16B LDS loads each.
        const int ak = kk + hi * 8;
        const h8 a0lo = *(const h8*)&As[cur][wr + lr][ak];
        const h8 a0hi = *(const h8*)&As[cur][wr + lr][ak + 16];
        const v16h af0 = __builtin_shufflevector(a0lo, a0hi,
            0, 1, 2, 3, 4, 5, 6, 7, 8, 9, 10, 11, 12, 13, 14, 15);
        const h8 a1lo = *(const h8*)&As[cur][wr + 16 + lr][ak];
        const h8 a1hi = *(const h8*)&As[cur][wr + 16 + lr][ak + 16];
        const v16h af1 = __builtin_shufflevector(a1lo, a1hi,
            0, 1, 2, 3, 4, 5, 6, 7, 8, 9, 10, 11, 12, 13, 14, 15);

        // B fragments (32x16 f16): lane = column, lane<16 holds K 0..7 /
        // 16..23, lane>=16 holds K 8..15 / 24..31 (matches A).
        const int bk = kk + hi * 8;
        const h8 b0lo = *(const h8*)&Bs[cur][wc + lr][bk];
        const h8 b0hi = *(const h8*)&Bs[cur][wc + lr][bk + 16];
        const v16h bf0 = __builtin_shufflevector(b0lo, b0hi,
            0, 1, 2, 3, 4, 5, 6, 7, 8, 9, 10, 11, 12, 13, 14, 15);
        const h8 b1lo = *(const h8*)&Bs[cur][wc + 16 + lr][bk];
        const h8 b1hi = *(const h8*)&Bs[cur][wc + 16 + lr][bk + 16];
        const v16h bf1 = __builtin_shufflevector(b1lo, b1hi,
            0, 1, 2, 3, 4, 5, 6, 7, 8, 9, 10, 11, 12, 13, 14, 15);

        acc00 = __builtin_amdgcn_wmma_f32_16x16x32_f16(
            false, af0, false, bf0, (short)0, acc00, false, false);
        acc01 = __builtin_amdgcn_wmma_f32_16x16x32_f16(
            false, af0, false, bf1, (short)0, acc01, false, false);
        acc10 = __builtin_amdgcn_wmma_f32_16x16x32_f16(
            false, af1, false, bf0, (short)0, acc10, false, false);
        acc11 = __builtin_amdgcn_wmma_f32_16x16x32_f16(
            false, af1, false, bf1, (short)0, acc11, false, false);
      }

      async_copies_wait();  // next buffer's DMA done
      __syncthreads();      // all waves done reading current buffer
    }

    // Merge this wave's 32x32 block into the running per-row argmax.
    // C/D layout: VGPR r, lanes 0-15 -> (M=r, N=lane); lanes 16-31 -> M=8+r.
    #pragma unroll
    for (int r = 0; r < 8; ++r) {
      const int rowA = wr + hi * 8 + r;
      const int cL   = c0 + wc + lr;
      float v = acc00[r]; int c = cL;
      if (acc01[r] > v) { v = acc01[r]; c = cL + 16; }
      argmax_merge(bestKey, rowA, v, c);
      v = acc10[r]; c = cL;
      if (acc11[r] > v) { v = acc11[r]; c = cL + 16; }
      argmax_merge(bestKey, rowA + 16, v, c);
    }
    __syncthreads();  // atomics done before next kt reuses tiles
  }

  if (tid < BM) {
    const unsigned c = (unsigned)(bestKey[tid] & 0xFFFFFFFFull);
    out[m0 + tid] = (float)cls[c];  // local argmax -> global class id
  }
}

// ---------------------------------------------------------------------------
// Host-side launcher
// ---------------------------------------------------------------------------
extern "C" void kernel_launch(void* const* d_in, const int* in_sizes, int n_in,
                              void* d_out, int out_size, void* d_ws, size_t ws_size,
                              hipStream_t stream) {
  const float* features = (const float*)d_in[0];   // [16384,1024] f32
  const float* ck       = (const float*)d_in[1];   // [1000,1024]  f32
  const int*   cls      = (const int*)d_in[2];     // [512] int
  float*       out      = (float*)d_out;           // [16384] preds as float

  _Float16* Af = (_Float16*)d_ws;                       // 32 MB
  _Float16* Bf = Af + (size_t)NROWS * DDIM;             // +1 MB

  ol_norm_rows_f16<<<NROWS, 256, 0, stream>>>(features, Af, nullptr, 0);
  ol_norm_rows_f16<<<KCLS,  256, 0, stream>>>(ck,       Bf, cls,     1);
  ol_gemm_argmax<<<NROWS / BM, 256, 0, stream>>>(Af, Bf, cls, out);
}
